// MPTAttention_2946347565281
// MI455X (gfx1250) — matrix operand order repaired and checked by reading, not compiled, not run on hardware
//
#include <hip/hip_runtime.h>
#include <hip/hip_bf16.h>

typedef __attribute__((ext_vector_type(16))) _Float16 v16h;
typedef __attribute__((ext_vector_type(8)))  _Float16 v8h;
typedef __attribute__((ext_vector_type(8)))  float    v8f;

static constexpr int BB = 2;
static constexpr int S  = 2048;
static constexpr int D  = 4096;
static constexpr int H  = 32;
static constexpr int HD = 128;          // head dim
static constexpr float SCALE = 0.08838834764831845f;  // HD^-0.5
static constexpr float CLIP  = 8.0f;

__device__ __forceinline__ v8f wmma16(v16h a, v16h b, v8f c) {
    // D = A(16x32 f16) * B(32x16 f16) + C(16x16 f32)
    return __builtin_amdgcn_wmma_f32_16x16x32_f16(false, a, false, b, (short)0, c, false, false);
}

// ---------------------------------------------------------------- fp32 -> f16
__global__ void cvt_f32_f16(const float* __restrict__ in, _Float16* __restrict__ out, long n) {
    long i = (long)blockIdx.x * blockDim.x + threadIdx.x;
    long stride = (long)gridDim.x * blockDim.x;
    for (; i < n; i += stride) out[i] = (_Float16)in[i];
}

// ---------------------------------------------------------------- WMMA GEMM
// C[M,N] = A[M,K] * B[K,N], f16 row-major inputs, f32 accumulate.
// MODE 0: store f32. MODE 1: clip to +/-8 and store f16.
// Block = 256 threads (8 waves, 4Mx2N) -> 128(M) x 64(N) tile.
// Wave tile = 32x32 (4 wmma per K-step). Software-pipelined global->LDS.
template <int MODE>
__global__ __launch_bounds__(256)
void gemm_wmma(const _Float16* __restrict__ A, const _Float16* __restrict__ Bm,
               float* __restrict__ Cf, _Float16* __restrict__ Ch,
               int M, int N, int K) {
    __shared__ _Float16 As[128][40];   // 128 rows x 32 k (+8 pad, rows 16B aligned)
    __shared__ _Float16 Bs[32][72];    // 32 k x 64 n (+8 pad)

    const int t    = threadIdx.x;
    const int lane = t & 31;
    const int wid  = t >> 5;
    const int hh   = lane >> 4;       // lane half
    const int n16  = lane & 15;
    const int wm   = wid >> 1;        // 0..3
    const int wn   = wid & 1;         // 0..1
    const int mBase = blockIdx.y * 128;
    const int nBase = blockIdx.x * 64;

    // staging assignments
    const int arow = t >> 1, aseg = (t & 1) * 16;   // A: 128x32, 16 halves/thread
    const int brow = t >> 3, bcol = (t & 7) * 8;    // B: 32x64,   8 halves/thread
    const _Float16* gA = A  + (size_t)(mBase + arow) * K + aseg;
    const _Float16* gB = Bm + (size_t)brow * N + nBase + bcol;

    v8f acc[4] = {};
    v8h aR0 = *(const v8h*)(gA);
    v8h aR1 = *(const v8h*)(gA + 8);
    v8h bR  = *(const v8h*)(gB);

    for (int kt = 0; kt < K; kt += 32) {
        *(v8h*)&As[arow][aseg]     = aR0;
        *(v8h*)&As[arow][aseg + 8] = aR1;
        *(v8h*)&Bs[brow][bcol]     = bR;
        __syncthreads();

        if (kt + 32 < K) {   // prefetch next K-step into registers (overlaps wmma)
            aR0 = *(const v8h*)(gA + kt + 32);
            aR1 = *(const v8h*)(gA + kt + 40);
            bR  = *(const v8h*)(gB + (size_t)(kt + 32) * N);
            __builtin_prefetch(gA + kt + 64, 0, 0);
            __builtin_prefetch(gB + (size_t)(kt + 64) * N, 0, 0);
        }

        // A fragments: rows wm*32 + {0,16} + n16 ; elem e -> k = e + 16*hh
        v16h a0, a1, b0, b1;
        {
            const _Float16* Ar0 = &As[wm * 32 + n16][0];
            const _Float16* Ar1 = &As[wm * 32 + 16 + n16][0];
            *((v8h*)&a0)     = *(const v8h*)(Ar0 + 16 * hh);
            *((v8h*)&a0 + 1) = *(const v8h*)(Ar0 + 8 + 16 * hh);
            *((v8h*)&a1)     = *(const v8h*)(Ar1 + 16 * hh);
            *((v8h*)&a1 + 1) = *(const v8h*)(Ar1 + 8 + 16 * hh);
        }
        // B fragments: cols wn*32 + {0,16} + n16 ; elem e -> k = e + 16*hh
#pragma unroll
        for (int e = 0; e < 16; ++e) {
            b0[e] = Bs[e + 16 * hh][wn * 32 + n16];
            b1[e] = Bs[e + 16 * hh][wn * 32 + 16 + n16];
        }

        acc[0] = wmma16(a0, b0, acc[0]);
        acc[1] = wmma16(a0, b1, acc[1]);
        acc[2] = wmma16(a1, b0, acc[2]);
        acc[3] = wmma16(a1, b1, acc[3]);
        __syncthreads();
    }

#pragma unroll
    for (int mi = 0; mi < 2; ++mi)
#pragma unroll
        for (int bi = 0; bi < 2; ++bi) {
            const v8f& acv = acc[mi * 2 + bi];
#pragma unroll
            for (int r = 0; r < 8; ++r) {
                const int row = mBase + wm * 32 + mi * 16 + r + 8 * hh;
                const int col = nBase + wn * 32 + bi * 16 + n16;
                float v = acv[r];
                if (MODE == 1) {
                    v = fminf(fmaxf(v, -CLIP), CLIP);
                    Ch[(size_t)row * N + col] = (_Float16)v;
                } else {
                    Cf[(size_t)row * N + col] = v;
                }
            }
        }
}

// ------------------------------------------------- LayerNorm(q,k) + relayout
// in : qkv f16 [B*S, 3D]   out: q/k/v f16 [B,H,S,HD]
__global__ __launch_bounds__(256)
void ln_split(const _Float16* __restrict__ qkv,
              const float* __restrict__ qw, const float* __restrict__ qb,
              const float* __restrict__ kw, const float* __restrict__ kb,
              _Float16* __restrict__ qO, _Float16* __restrict__ kO, _Float16* __restrict__ vO) {
    const int bs = blockIdx.x;          // 0..B*S-1
    const int b  = bs / S, s = bs % S;
    const int t  = threadIdx.x;
    const int lane = t & 31, wid = t >> 5;
    const _Float16* row = qkv + (size_t)bs * (3 * D);
    __shared__ float redS[16];

#pragma unroll
    for (int part = 0; part < 2; ++part) {
        const _Float16* p = row + part * D;
        float x[16];
        float sum = 0.f, sq = 0.f;
#pragma unroll
        for (int i = 0; i < 16; ++i) {
            const int c = t + i * 256;
            x[i] = (float)p[c];
            sum += x[i];
            sq  += x[i] * x[i];
        }
#pragma unroll
        for (int off = 1; off < 32; off <<= 1) {
            sum += __shfl_xor(sum, off, 32);
            sq  += __shfl_xor(sq,  off, 32);
        }
        __syncthreads();
        if (lane == 0) { redS[wid] = sum; redS[8 + wid] = sq; }
        __syncthreads();
        float tsum = 0.f, tsq = 0.f;
#pragma unroll
        for (int w = 0; w < 8; ++w) { tsum += redS[w]; tsq += redS[8 + w]; }
        const float mu   = tsum / (float)D;
        const float var  = tsq / (float)D - mu * mu;
        const float rstd = rsqrtf(var + 1e-5f);
        const float* w  = part ? kw : qw;
        const float* bb = part ? kb : qb;
        _Float16* out   = part ? kO : qO;
#pragma unroll
        for (int i = 0; i < 16; ++i) {
            const int c = t + i * 256;
            const int head = c >> 7, d = c & 127;
            out[(((size_t)b * H + head) * S + s) * HD + d] =
                (_Float16)((x[i] - mu) * rstd * w[c] + bb[c]);
        }
    }
#pragma unroll
    for (int i = 0; i < 16; ++i) {
        const int c = t + i * 256;
        const int head = c >> 7, d = c & 127;
        vO[(((size_t)b * H + head) * S + s) * HD + d] = row[2 * D + c];
    }
}

// ------------------------------------------------ V transpose: [BH,S,HD] -> [BH,HD,S]
__global__ __launch_bounds__(256)
void transpose_v(const _Float16* __restrict__ in, _Float16* __restrict__ out) {
    const int bh = blockIdx.z;
    const int s0 = blockIdx.x * 64;
    const int d0 = blockIdx.y * 64;
    __shared__ _Float16 tile[64][72];
    const int t    = threadIdx.x;
    const int r    = t >> 2;           // 0..63
    const int cseg = (t & 3) * 16;     // 0,16,32,48

    const _Float16* src = in + ((size_t)bh * S + s0 + r) * HD + d0 + cseg;
    *(v8h*)&tile[r][cseg]     = *(const v8h*)(src);
    *(v8h*)&tile[r][cseg + 8] = *(const v8h*)(src + 8);
    __syncthreads();

    v8h o0, o1;
#pragma unroll
    for (int j = 0; j < 8; ++j) {
        o0[j] = tile[cseg + j][r];
        o1[j] = tile[cseg + 8 + j][r];
    }
    _Float16* dst = out + ((size_t)bh * HD + d0 + r) * S + s0 + cseg;
    *(v8h*)(dst)     = o0;
    *(v8h*)(dst + 8) = o1;
}

// ------------------------------------------------------ flash attention wave
// q/k: f16 [B,H,S,HD].  vT: f16 [B,H,HD,S].  ao: f16 [B*S, D] (col = h*HD+d).
// One wave per 16-query tile. Scores computed transposed (S^T = K * Q^T) so
// score C-fragments feed the P*V A-fragment lane-for-lane.
__global__ __launch_bounds__(32)
void attn_wmma(const _Float16* __restrict__ qH, const _Float16* __restrict__ kH,
               const _Float16* __restrict__ vT, _Float16* __restrict__ ao) {
    const int qt = blockIdx.x, h = blockIdx.y, b = blockIdx.z;
    const int qbase = qt * 16;
    const int lane = threadIdx.x;
    const int hh  = lane >> 4;
    const int n16 = lane & 15;
    const float slope = exp2f(-0.25f * (float)(h + 1));

    const size_t headOff = ((size_t)b * H + h) * S * (size_t)HD;

    // Q^T B-fragments (reused every chunk): elem e -> d = dc*32 + 16*hh + e
    v16h qf[4];
    {
        const _Float16* qRow = qH + headOff + (size_t)(qbase + n16) * HD;
#pragma unroll
        for (int dc = 0; dc < 4; ++dc) {
            const int off = dc * 32 + 16 * hh;
            *((v8h*)&qf[dc])     = *(const v8h*)(qRow + off);
            *((v8h*)&qf[dc] + 1) = *(const v8h*)(qRow + off + 8);
        }
    }

    v8f o[8] = {};          // O[16q x 128d], lane = d-col, vgpr = q-row
    float mrun = -INFINITY; // stats for query (lane&15)
    float lrun = 0.f;

    const int nchunks = (qbase + 16 + 31) / 32;
    const int qgi = qbase + n16;
    const _Float16* vtHead = vT + headOff;   // [HD][S]

    for (int c = 0; c < nchunks; ++c) {
        const int kbase = c * 32;
        // S^T tiles: A = K-tile (16 keys x 32 d), B = Q^T. Lane m = key.
        v8f c0 = {}, c1 = {};
        const _Float16* kRow0 = kH + headOff + (size_t)(kbase + n16) * HD;
        const _Float16* kRow1 = kRow0 + 16 * HD;
#pragma unroll
        for (int dc = 0; dc < 4; ++dc) {
            const int off = dc * 32 + 16 * hh;
            v16h a0, a1;
            *((v8h*)&a0)     = *(const v8h*)(kRow0 + off);
            *((v8h*)&a0 + 1) = *(const v8h*)(kRow0 + off + 8);
            *((v8h*)&a1)     = *(const v8h*)(kRow1 + off);
            *((v8h*)&a1 + 1) = *(const v8h*)(kRow1 + off + 8);
            c0 = wmma16(a0, qf[dc], c0);
            c1 = wmma16(a1, qf[dc], c1);
        }

        // scale + ALiBi + causal mask (lane holds keys r+8*hh for query qgi)
        float s0[8], s1[8];
#pragma unroll
        for (int r = 0; r < 8; ++r) {
            const int k0 = kbase + r + 8 * hh;
            const int k1 = k0 + 16;
            s0[r] = (k0 <= qgi) ? c0[r] * SCALE - slope * (float)(qgi - k0) : -INFINITY;
            s1[r] = (k1 <= qgi) ? c1[r] * SCALE - slope * (float)(qgi - k1) : -INFINITY;
        }
        float cm = -INFINITY;
#pragma unroll
        for (int r = 0; r < 8; ++r) cm = fmaxf(cm, fmaxf(s0[r], s1[r]));
        cm = fmaxf(cm, __shfl_xor(cm, 16, 32));
        const float mnew  = fmaxf(mrun, cm);
        const float alpha = __expf(mrun - mnew);

        // P as A-fragment: elem e<8 -> tile0 row e ; e>=8 -> tile1 row e-8
        v16h pa;
        float ps = 0.f;
#pragma unroll
        for (int r = 0; r < 8; ++r) {
            const float p0 = __expf(s0[r] - mnew);
            const float p1 = __expf(s1[r] - mnew);
            ps += p0 + p1;
            pa[r]     = (_Float16)p0;
            pa[r + 8] = (_Float16)p1;
        }
        ps += __shfl_xor(ps, 16, 32);
        lrun = lrun * alpha + ps;
        mrun = mnew;

        // rescale O: row r+8*hh uses alpha of query r+8*hh
        float arow[8];
#pragma unroll
        for (int r = 0; r < 8; ++r) arow[r] = __shfl(alpha, r + 8 * hh, 32);
#pragma unroll
        for (int dt = 0; dt < 8; ++dt)
#pragma unroll
            for (int r = 0; r < 8; ++r) o[dt][r] *= arow[r];

        // O += P * V ; V B-fragment from vT: elem e -> key = kbase+(e<8?e:e+8)+8*hh
#pragma unroll
        for (int dt = 0; dt < 8; ++dt) {
            const _Float16* vr = vtHead + (size_t)(dt * 16 + n16) * S + kbase + 8 * hh;
            v16h vb;
            *((v8h*)&vb)     = *(const v8h*)(vr);
            *((v8h*)&vb + 1) = *(const v8h*)(vr + 16);
            o[dt] = wmma16(pa, vb, o[dt]);
        }
    }

    const float linv = 1.f / lrun;
    float lr[8];
#pragma unroll
    for (int r = 0; r < 8; ++r) lr[r] = __shfl(linv, r + 8 * hh, 32);
#pragma unroll
    for (int dt = 0; dt < 8; ++dt)
#pragma unroll
        for (int r = 0; r < 8; ++r) {
            const size_t row = (size_t)b * S + qbase + r + 8 * hh;
            ao[row * (size_t)D + h * HD + dt * 16 + n16] = (_Float16)(o[dt][r] * lr[r]);
        }
}

// ---------------------------------------------------------------------------
extern "C" void kernel_launch(void* const* d_in, const int* in_sizes, int n_in,
                              void* d_out, int out_size, void* d_ws, size_t ws_size,
                              hipStream_t stream) {
    const float* hs   = (const float*)d_in[0];
    const float* Wqkv = (const float*)d_in[1];
    const float* Wout = (const float*)d_in[2];
    const float* qw   = (const float*)d_in[3];
    const float* qb   = (const float*)d_in[4];
    const float* kw   = (const float*)d_in[5];
    const float* kb   = (const float*)d_in[6];
    float* out = (float*)d_out;

    _Float16* ws = (_Float16*)d_ws;
    size_t off = 0;
    _Float16* hsH   = ws + off; off += (size_t)BB * S * D;
    _Float16* wqkvH = ws + off; off += (size_t)D * 3 * D;
    _Float16* woutH = ws + off; off += (size_t)D * D;
    _Float16* qkvH  = ws + off; off += (size_t)BB * S * 3 * D;
    _Float16* qHh   = ws + off; off += (size_t)BB * S * D;
    _Float16* kHh   = ws + off; off += (size_t)BB * S * D;
    _Float16* vHh   = ws + off; off += (size_t)BB * S * D;
    _Float16* vTh   = ws + off; off += (size_t)BB * S * D;
    _Float16* aoH   = ws + off; off += (size_t)BB * S * D;
    (void)ws_size; (void)in_sizes; (void)n_in; (void)out_size;

    cvt_f32_f16<<<8192, 256, 0, stream>>>(hs,   hsH,   (long)BB * S * D);
    cvt_f32_f16<<<8192, 256, 0, stream>>>(Wqkv, wqkvH, (long)D * 3 * D);
    cvt_f32_f16<<<8192, 256, 0, stream>>>(Wout, woutH, (long)D * D);

    // QKV = clip(hs @ Wqkv): M=4096, N=12288, K=4096
    gemm_wmma<1><<<dim3(3 * D / 64, BB * S / 128), 256, 0, stream>>>(
        hsH, wqkvH, nullptr, qkvH, BB * S, 3 * D, D);

    ln_split<<<BB * S, 256, 0, stream>>>(qkvH, qw, qb, kw, kb, qHh, kHh, vHh);

    transpose_v<<<dim3(S / 64, HD / 64, BB * H), 256, 0, stream>>>(vHh, vTh);

    attn_wmma<<<dim3(S / 16, H, BB), 32, 0, stream>>>(qHh, kHh, vTh, aoH);

    // out = ao @ Wout: M=4096, N=4096, K=4096, f32 epilogue
    gemm_wmma<0><<<dim3(D / 64, BB * S / 128), 256, 0, stream>>>(
        aoH, woutH, out, nullptr, BB * S, D, D);
}
